// AutoregressivePPRM_46737834115554
// MI455X (gfx1250) — matrix operand order repaired and checked by reading, not compiled, not run on hardware
//
#include <hip/hip_runtime.h>

#define B_   32
#define N_   500
#define D_   256
#define HOR  24
#define BOT  8
#define BN   (B_*N_)
#define HSTR 264   // LDS h row stride in bf16: 528B = 16B aligned, conflict-free

typedef __attribute__((ext_vector_type(16))) __bf16 v16bf;
typedef __attribute__((ext_vector_type(8)))  __bf16 v8bf;
typedef __attribute__((ext_vector_type(8)))  float  v8f;

union AFrag { v16bf v; v8bf h[2]; };

__device__ __forceinline__ unsigned short f2bf(float f) {
    unsigned u = __float_as_uint(f);
    u += 0x7FFFu + ((u >> 16) & 1u);           // round-to-nearest-even
    return (unsigned short)(u >> 16);
}
__device__ __forceinline__ float bf2f(unsigned short s) {
    return __uint_as_float(((unsigned)s) << 16);
}
__device__ __forceinline__ float sigm(float x) { return 1.0f / (1.0f + __expf(-x)); }
__device__ __forceinline__ float tanh_(float x) {
    x = fminf(fmaxf(x, -15.f), 15.f);
    float e = __expf(2.0f * x);
    return (e - 1.0f) / (e + 1.0f);
}

// One-time fp32 -> bf16 conversion of W_hh into workspace (L2-resident, 393KB)
__global__ void cvt_bf16(const float* __restrict__ src, unsigned short* __restrict__ dst, int n) {
    int i = blockIdx.x * blockDim.x + threadIdx.x;
    if (i < n) dst[i] = f2bf(src[i]);
}

__global__ __launch_bounds__(256) void gru_fused(
    const float* __restrict__ features, const float* __restrict__ last_value,
    const float* __restrict__ W_ih, const float* __restrict__ b_ih,
    const float* __restrict__ b_hh,
    const float* __restrict__ Wl, const float* __restrict__ bl,
    const float* __restrict__ Wh, const float* __restrict__ bh,
    const float* __restrict__ Wg1, const float* __restrict__ bg1,
    const float* __restrict__ Wg2, const float* __restrict__ bg2,
    const float* __restrict__ log_decay,
    const unsigned short* __restrict__ wbf,     // W_hh bf16 [768][256] row-major
    float* __restrict__ out)
{
    __shared__ __align__(16) unsigned short hbuf[16 * HSTR]; // h tile, bf16
    __shared__ float wl_lds[BOT * D_];
    __shared__ float psum[256];
    __shared__ float p8_lds[16 * BOT];
    __shared__ float g1_lds[16 * BOT];
    __shared__ float gate_lds[16 * HOR];
    __shared__ float x_lds[16];
    __shared__ float lv_lds[16];

    const int tid    = threadIdx.x;
    const int row0   = blockIdx.x * 16;
    const int w      = tid >> 5;      // wave id 0..7
    const int lane   = tid & 31;
    const int laneLo = lane & 15;
    const int laneHi = lane >> 4;

    // Per-wave gate-column constants (invariant over all 24 steps) and h in regs.
    // Wave w owns hidden column groups g = w and w+8 (16 cols each).
    int   colH[2];
    float wih[2][3], bih[2][3], bhh[2][3];
    float hold[2][8];
    #pragma unroll
    for (int i = 0; i < 2; i++) {
        int g = w + 8 * i;
        colH[i] = 16 * g + laneLo;
        #pragma unroll
        for (int q = 0; q < 3; q++) {            // q: 0=r, 1=z, 2=n  (offsets 0/256/512)
            wih[i][q] = W_ih[q * 256 + colH[i]];
            bih[i][q] = b_ih[q * 256 + colH[i]];
            bhh[i][q] = b_hh[q * 256 + colH[i]];
        }
        #pragma unroll
        for (int r = 0; r < 8; r++)              // C-layout: m = r + 8*laneHi
            hold[i][r] = features[(row0 + r + laneHi * 8) * D_ + colH[i]];
    }

    // Stage h0 = features (bf16) and Wl into LDS; init x = last_value.
    for (int idx = tid; idx < 16 * D_; idx += 256) {
        int r = idx >> 8, c = idx & 255;
        hbuf[r * HSTR + c] = f2bf(features[(row0 + r) * D_ + c]);
    }
    for (int idx = tid; idx < BOT * D_; idx += 256) wl_lds[idx] = Wl[idx];
    if (tid < 16) {
        float lv = last_value[row0 + tid];
        lv_lds[tid] = lv;
        x_lds[tid]  = lv;
    }
    __syncthreads();

    // gate = sigmoid(relu(features @ Wg1^T + bg1) @ Wg2^T + bg2), once per block.
    {
        int row = tid >> 4, bot = (tid >> 1) & 7, half = tid & 1;
        const float* fp = &features[(row0 + row) * D_ + half * 128];
        const float* wp = &Wg1[bot * D_ + half * 128];
        float p = 0.f;
        for (int c = 0; c < 128; c++) p = fmaf(fp[c], wp[c], p);
        psum[tid] = p;
    }
    __syncthreads();
    if (tid < 128) {
        int row = tid >> 3, bot = tid & 7;
        float v = psum[row * 16 + bot * 2] + psum[row * 16 + bot * 2 + 1] + bg1[bot];
        g1_lds[tid] = fmaxf(v, 0.f);
    }
    __syncthreads();
    for (int idx = tid; idx < 16 * HOR; idx += 256) {
        int row = idx / HOR, s = idx % HOR;
        float a = bg2[s];
        #pragma unroll
        for (int b2 = 0; b2 < 8; b2++) a = fmaf(Wg2[s * 8 + b2], g1_lds[row * 8 + b2], a);
        gate_lds[idx] = sigm(a);
    }
    __syncthreads();

    const float decay_rate = __expf(log_decay[0]);

    for (int s = 0; s < HOR; s++) {
        v8f zero = {};
        v8f acc[6] = { zero, zero, zero, zero, zero, zero };

        // gh = h @ W_hh^T : A = h tile (LDS bf16), B = W_hh slices (L2 bf16).
        // acc[i] covers N-tile (w + 8*i): i=0,1 -> r gate, 2,3 -> z, 4,5 -> n.
        #pragma unroll
        for (int kk = 0; kk < 8; kk++) {
            AFrag a;
            const unsigned short* ap = &hbuf[laneLo * HSTR + kk * 32 + laneHi * 16];
            a.h[0] = *(const v8bf*)(ap);        // K 0..7  (or 16..23 for hi lanes)
            a.h[1] = *(const v8bf*)(ap + 8);    // K 8..15 (or 24..31)
            #pragma unroll
            for (int i = 0; i < 6; i++) {
                int nt = w + 8 * i;
                const v16bf bfr = *(const v16bf*)(wbf + (16 * nt + laneLo) * D_
                                                  + kk * 32 + laneHi * 16);
                acc[i] = __builtin_amdgcn_wmma_f32_16x16x32_bf16(
                             false, a.v, false, bfr, (short)0, acc[i], false, false);
            }
        }

        // GRU elementwise update, fully in registers (r/z/n lane-aligned).
        float xs[8];
        #pragma unroll
        for (int r = 0; r < 8; r++) xs[r] = x_lds[r + laneHi * 8];
        __syncthreads();   // all A-fragment reads of hbuf done before overwrite
        #pragma unroll
        for (int g = 0; g < 2; g++) {
            #pragma unroll
            for (int r = 0; r < 8; r++) {
                float hr = acc[g][r]     + bhh[g][0];
                float hz = acc[2 + g][r] + bhh[g][1];
                float hn = acc[4 + g][r] + bhh[g][2];
                float rr = sigm(fmaf(xs[r], wih[g][0], bih[g][0]) + hr);
                float zz = sigm(fmaf(xs[r], wih[g][1], bih[g][1]) + hz);
                float nc = tanh_(fmaf(xs[r], wih[g][2], bih[g][2]) + rr * hn);
                float hv = (1.f - zz) * nc + zz * hold[g][r];
                hold[g][r] = hv;
                hbuf[(r + laneHi * 8) * HSTR + colH[g]] = f2bf(hv);
            }
        }
        __syncthreads();   // h_new visible to all waves

        // pred = (h_new @ Wl^T + bl) @ Wh^T + bh  (split-K over 256 threads)
        {
            int row = tid >> 4, bot = (tid >> 1) & 7, half = tid & 1;
            const unsigned short* hp = &hbuf[row * HSTR + half * 128];
            const float* wp = &wl_lds[bot * D_ + half * 128];
            float p = 0.f;
            for (int c = 0; c < 128; c++) p = fmaf(bf2f(hp[c]), wp[c], p);
            psum[tid] = p;
        }
        __syncthreads();
        if (tid < 128) {
            int row = tid >> 3, bot = tid & 7;
            p8_lds[tid] = psum[row * 16 + bot * 2] + psum[row * 16 + bot * 2 + 1] + bl[bot];
        }
        __syncthreads();
        if (tid < 16) {
            float p = bh[0];
            #pragma unroll
            for (int b2 = 0; b2 < 8; b2++) p = fmaf(p8_lds[tid * 8 + b2], Wh[b2], p);
            float gv  = gate_lds[tid * HOR + s];
            float dec = lv_lds[tid] * __expf(-decay_rate * (float)(s + 1));
            out[(row0 + tid) * HOR + s] = gv * p + (1.f - gv) * dec;
            x_lds[tid] = p;    // next step's GRU input
        }
        __syncthreads();
    }
}

extern "C" void kernel_launch(void* const* d_in, const int* in_sizes, int n_in,
                              void* d_out, int out_size, void* d_ws, size_t ws_size,
                              hipStream_t stream)
{
    const float* features   = (const float*)d_in[0];
    const float* last_value = (const float*)d_in[1];
    const float* W_ih       = (const float*)d_in[2];
    const float* W_hh       = (const float*)d_in[3];
    const float* b_ih       = (const float*)d_in[4];
    const float* b_hh       = (const float*)d_in[5];
    const float* Wl         = (const float*)d_in[6];
    const float* bl         = (const float*)d_in[7];
    const float* Wh         = (const float*)d_in[8];
    const float* bh         = (const float*)d_in[9];
    const float* Wg1        = (const float*)d_in[10];
    const float* bg1        = (const float*)d_in[11];
    const float* Wg2        = (const float*)d_in[12];
    const float* bg2        = (const float*)d_in[13];
    const float* log_decay  = (const float*)d_in[14];

    unsigned short* wbf = (unsigned short*)d_ws;   // 3*D*D bf16 = 393216 bytes
    const int nW = 3 * D_ * D_;
    cvt_bf16<<<(nW + 255) / 256, 256, 0, stream>>>(W_hh, wbf, nW);

    gru_fused<<<BN / 16, 256, 0, stream>>>(features, last_value, W_ih, b_ih, b_hh,
                                           Wl, bl, Wh, bh, Wg1, bg1, Wg2, bg2,
                                           log_decay, wbf, (float*)d_out);
}
